// Attention_47029891891630
// MI455X (gfx1250) — matrix operand order repaired and checked by reading, not compiled, not run on hardware
//
#include <hip/hip_runtime.h>
#include <hip/hip_bf16.h>
#include <math.h>

typedef __attribute__((ext_vector_type(16))) _Float16 v16h;
typedef __attribute__((ext_vector_type(8)))  _Float16 v8h;
typedef __attribute__((ext_vector_type(8)))  float    v8f;

union H16 { v16h v; _Float16 h[16]; v8h p[2]; };
union H8  { v8h  v; _Float16 h[8]; };

// ---------------------------------------------------------------------------
// Fragment loaders (wave32; contiguous k order in A and B alike, not the ISA table order)
// A-matrix 16x32 f16: lane m=lane&15, hi=lane>>4 holds K = hi*16..+15
// of row m  -> two contiguous 16B loads.
// B-matrix 32x16 f16: lane (n=lane&15, hi) holds column n for K=16*hi..+15,
// i.e. 32 contiguous bytes of row n of B^T.
// ---------------------------------------------------------------------------
__device__ __forceinline__ v16h load_a_f16(const _Float16* __restrict__ base,
                                           int ld, int lane) {
  const int m = lane & 15, hi = lane >> 4;
  const _Float16* p = base + (size_t)m * ld + hi * 16;
  H16 r;
  r.p[0] = *(const v8h*)(p);
  r.p[1] = *(const v8h*)(p + 8);
  return r.v;
}

__device__ __forceinline__ v16h load_a_f32(const float* __restrict__ base,
                                           int ld, int lane) {
  const int m = lane & 15, hi = lane >> 4;
  const float* p = base + (size_t)m * ld + hi * 16;
  H16 r;
#pragma unroll
  for (int i = 0; i < 8; ++i) r.h[i] = (_Float16)p[i];
#pragma unroll
  for (int i = 0; i < 8; ++i) r.h[8 + i] = (_Float16)p[8 + i];
  return r.v;
}

__device__ __forceinline__ v16h load_bt_f16(const _Float16* __restrict__ baseT,
                                            int ld, int lane) {
  const int n = lane & 15, hi = lane >> 4;
  return *(const v16h*)(baseT + (size_t)n * ld + hi * 16);
}

// C/D 16x16 f32 layout: lane (n=lane&15, chi=lane>>4), VGPR r -> M = r + 8*chi
__device__ __forceinline__ void store_c_f32(float* __restrict__ Cb, int ldc,
                                            int mbase, int nbase, const v8f& acc,
                                            int cn, int chi, float scale) {
  const int nn = nbase + cn;
  const int mb = mbase + chi * 8;
#pragma unroll
  for (int r = 0; r < 8; ++r)
    Cb[(size_t)(mb + r) * ldc + nn] = acc[r] * scale;
}

__device__ __forceinline__ void store_c_f16rm(_Float16* __restrict__ Cb, int ldc,
                                              int mbase, int nbase, const v8f& acc,
                                              int cn, int chi, float bval) {
  const int nn = nbase + cn;
  const int mb = mbase + chi * 8;
#pragma unroll
  for (int r = 0; r < 8; ++r)
    Cb[(size_t)(mb + r) * ldc + nn] = (_Float16)(acc[r] + bval);
}

// Transposed f16 store: outT is [B][N][S]; lane's 8 consecutive M values land
// contiguously in row nn -> one packed 16-byte store.
__device__ __forceinline__ void store_c_f16t(_Float16* __restrict__ outT, int S,
                                             int N, int mbase, int nbase,
                                             const v8f& acc, int cn, int chi,
                                             float bval) {
  const int nn = nbase + cn;
  const int mb = mbase + chi * 8;
  const int b = mb / S, s = mb % S;
  H8 pk;
#pragma unroll
  for (int r = 0; r < 8; ++r) pk.h[r] = (_Float16)(acc[r] + bval);
  *(v8h*)(outT + ((size_t)b * N + nn) * S + s) = pk.v;
}

#define WMMA_F16(A, B, C) \
  __builtin_amdgcn_wmma_f32_16x16x32_f16(false, (A), false, (B), (short)0, (C), false, false)

// ---------------------------------------------------------------------------
// f32 -> f16 elementwise convert (weights)
// ---------------------------------------------------------------------------
__global__ void cvt_f32_to_f16(const float* __restrict__ in,
                               _Float16* __restrict__ out, int n) {
  int i = blockIdx.x * blockDim.x + threadIdx.x;
  if (i < n) out[i] = (_Float16)in[i];
}

// ---------------------------------------------------------------------------
// Projection: C[M,N] = X[M,K](f32) * W^T (W16 is [N,K] f16) + bias
// Block = 256 thr = 8 waves (4x2), wave tile 32x32 -> block tile 128x64.
// outRM != nullptr : store f16 row-major [M,N]
// else             : store f16 transposed per-batch [B][N][S]
// ---------------------------------------------------------------------------
__global__ __launch_bounds__(256) void proj_gemm(
    const float* __restrict__ X, const _Float16* __restrict__ W16,
    const float* __restrict__ bias, _Float16* __restrict__ outRM,
    _Float16* __restrict__ outT, int K, int N, int S) {
  const int lane = threadIdx.x & 31;
  const int wave = threadIdx.x >> 5;
  const int m0 = blockIdx.x * 128 + (wave & 3) * 32;
  const int n0 = blockIdx.y * 64 + (wave >> 2) * 32;

  v8f acc00 = {}, acc01 = {}, acc10 = {}, acc11 = {};
  for (int k = 0; k < K; k += 32) {
    v16h a0 = load_a_f32(X + (size_t)m0 * K + k, K, lane);
    v16h a1 = load_a_f32(X + (size_t)(m0 + 16) * K + k, K, lane);
    v16h b0 = load_bt_f16(W16 + (size_t)n0 * K + k, K, lane);
    v16h b1 = load_bt_f16(W16 + (size_t)(n0 + 16) * K + k, K, lane);
    acc00 = WMMA_F16(a0, b0, acc00);
    acc01 = WMMA_F16(a0, b1, acc01);
    acc10 = WMMA_F16(a1, b0, acc10);
    acc11 = WMMA_F16(a1, b1, acc11);
  }

  const int cn = lane & 15, chi = lane >> 4;
  const float bv0 = bias[n0 + cn];
  const float bv1 = bias[n0 + 16 + cn];
  if (outRM) {
    store_c_f16rm(outRM, N, m0,      n0,      acc00, cn, chi, bv0);
    store_c_f16rm(outRM, N, m0,      n0 + 16, acc01, cn, chi, bv1);
    store_c_f16rm(outRM, N, m0 + 16, n0,      acc10, cn, chi, bv0);
    store_c_f16rm(outRM, N, m0 + 16, n0 + 16, acc11, cn, chi, bv1);
  } else {
    store_c_f16t(outT, S, N, m0,      n0,      acc00, cn, chi, bv0);
    store_c_f16t(outT, S, N, m0,      n0 + 16, acc01, cn, chi, bv1);
    store_c_f16t(outT, S, N, m0 + 16, n0,      acc10, cn, chi, bv0);
    store_c_f16t(outT, S, N, m0 + 16, n0 + 16, acc11, cn, chi, bv1);
  }
}

// ---------------------------------------------------------------------------
// Batched C = scale * A(f16,[M,K]) * Bt(f16,[N,K])^T, C f32 [M,N].
// Block = 256 thr = 8 waves; block tile 256(M) x 64(N); wave tile 32x64
// (8 accumulators -> 8 independent WMMAs per K-step).
// The 64x32-half B tile (4KB) shared by all 8 waves is staged into LDS with
// GLOBAL_LOAD_ASYNC_TO_LDS_B128 (one 16B chunk per thread, ASYNCcnt), DOUBLE
// BUFFERED: tile i+1 streams into the other LDS buffer while tile i is
// consumed by WMMAs; s_wait_asynccnt 1 retires only the older transfer
// (async loads complete in order, ISA 10.4.1).
// LDS row pitch 80B keeps the 16-lane ds_load_b128 pattern conflict-free.
// ---------------------------------------------------------------------------
#define BT_PITCH_H 40                    // halves per LDS row (80 bytes)
#define BT_TILE_H  (64 * BT_PITCH_H)     // halves per buffer

__device__ __forceinline__ void stage_tile_async(const _Float16* gsrc,
                                                 unsigned lds_dst) {
  unsigned long long ga = (unsigned long long)(uintptr_t)gsrc;
  asm volatile("global_load_async_to_lds_b128 %0, %1, off"
               :: "v"(lds_dst), "v"(ga)
               : "memory");
}

__device__ __forceinline__ void compute_tile(const _Float16* btile, v16h a0,
                                             v16h a1, v8f (&acc)[2][4],
                                             int lane) {
#pragma unroll
  for (int j = 0; j < 4; ++j) {
    const _Float16* bp =
        btile + (j * 16 + (lane & 15)) * BT_PITCH_H + (lane >> 4) * 16;
    H16 b;
    b.p[0] = *(const v8h*)(bp);
    b.p[1] = *(const v8h*)(bp + 8);
    acc[0][j] = WMMA_F16(a0, b.v, acc[0][j]);
    acc[1][j] = WMMA_F16(a1, b.v, acc[1][j]);
  }
}

__global__ __launch_bounds__(256) void gemm_a16_bt16(
    const _Float16* __restrict__ A, size_t sAb, int lda,
    const _Float16* __restrict__ Bt, size_t sBb, int ldb,
    float* __restrict__ C, size_t sCb, int ldc, int K, float scale) {
  __shared__ _Float16 btile[2 * BT_TILE_H];

  const int lane = threadIdx.x & 31;
  const int wave = threadIdx.x >> 5;
  const int m0 = blockIdx.x * 256 + wave * 32;
  const int n0 = blockIdx.y * 64;
  const _Float16* Ab = A + (size_t)blockIdx.z * sAb;
  const _Float16* Btb = Bt + (size_t)blockIdx.z * sBb;
  float* Cb = C + (size_t)blockIdx.z * sCb;

  // Per-thread async-copy slot: thread t moves 16B of row (t>>2), chunk (t&3)
  const int trow = threadIdx.x >> 2;
  const int tchk = threadIdx.x & 3;
  const _Float16* gsrc = Btb + (size_t)(n0 + trow) * ldb + tchk * 8;
  const unsigned lds0 =
      (unsigned)(uintptr_t)(btile + trow * BT_PITCH_H + tchk * 8);
  const unsigned lds1 = lds0 + BT_TILE_H * 2 /*bytes*/;

  const _Float16* a0p = Ab + (size_t)m0 * lda;
  const _Float16* a1p = Ab + (size_t)(m0 + 16) * lda;

  v8f acc[2][4] = {};

  // Prologue: tile 0 -> buffer 0
  stage_tile_async(gsrc, lds0);

  for (int k = 0; k < K; k += 64) {
    // ---- even tile (buffer 0) ----
    v16h a0 = load_a_f16(a0p + k, lda, lane);
    v16h a1 = load_a_f16(a1p + k, lda, lane);
    if (k + 32 < K) {
      stage_tile_async(gsrc + k + 32, lds1);  // prefetch tile k+32 -> buf1
      asm volatile("s_wait_asynccnt 0x1" ::: "memory");
    } else {
      asm volatile("s_wait_asynccnt 0x0" ::: "memory");
    }
    __syncthreads();  // buffer 0 visible to all waves
    compute_tile(btile, a0, a1, acc, lane);
    __syncthreads();  // all waves done reading buffer 0

    if (k + 32 >= K) break;

    // ---- odd tile (buffer 1) ----
    a0 = load_a_f16(a0p + k + 32, lda, lane);
    a1 = load_a_f16(a1p + k + 32, lda, lane);
    if (k + 64 < K) {
      stage_tile_async(gsrc + k + 64, lds0);  // prefetch tile k+64 -> buf0
      asm volatile("s_wait_asynccnt 0x1" ::: "memory");
    } else {
      asm volatile("s_wait_asynccnt 0x0" ::: "memory");
    }
    __syncthreads();  // buffer 1 visible to all waves
    compute_tile(btile + BT_TILE_H, a0, a1, acc, lane);
    __syncthreads();  // all waves done reading buffer 1
  }

  const int cn = lane & 15, chi = lane >> 4;
#pragma unroll
  for (int j = 0; j < 4; ++j) {
    store_c_f32(Cb, ldc, m0,      n0 + 16 * j, acc[0][j], cn, chi, scale);
    store_c_f32(Cb, ldc, m0 + 16, n0 + 16 * j, acc[1][j], cn, chi, scale);
  }
}

// ---------------------------------------------------------------------------
// Row softmax: scores f32 [rows, S] -> probs f16 [rows, S]. One 256-thr block
// per row, S = 2048 (8 elements/thread).
// ---------------------------------------------------------------------------
__global__ __launch_bounds__(256) void softmax_rows(
    const float* __restrict__ scores, _Float16* __restrict__ probs, int S) {
  __shared__ float red[256];
  const int t = threadIdx.x;
  const float* src = scores + (size_t)blockIdx.x * S;
  _Float16* dst = probs + (size_t)blockIdx.x * S;

  float vals[8];
  float m = -1e30f;
#pragma unroll
  for (int i = 0; i < 8; ++i) {
    vals[i] = src[t + i * 256];
    m = fmaxf(m, vals[i]);
  }
  red[t] = m;
  __syncthreads();
  for (int off = 128; off > 0; off >>= 1) {
    if (t < off) red[t] = fmaxf(red[t], red[t + off]);
    __syncthreads();
  }
  m = red[0];
  __syncthreads();

  float s = 0.f;
#pragma unroll
  for (int i = 0; i < 8; ++i) {
    vals[i] = __expf(vals[i] - m);
    s += vals[i];
  }
  red[t] = s;
  __syncthreads();
  for (int off = 128; off > 0; off >>= 1) {
    if (t < off) red[t] += red[t + off];
    __syncthreads();
  }
  const float inv = 1.f / red[0];
#pragma unroll
  for (int i = 0; i < 8; ++i) dst[t + i * 256] = (_Float16)(vals[i] * inv);
}

// ---------------------------------------------------------------------------
extern "C" void kernel_launch(void* const* d_in, const int* in_sizes, int n_in,
                              void* d_out, int out_size, void* d_ws,
                              size_t ws_size, hipStream_t stream) {
  (void)in_sizes; (void)n_in; (void)out_size; (void)ws_size;
  constexpr int B = 4, S = 2048, D = 768;
  constexpr int M = B * S;  // 8192 flattened rows

  const float* q  = (const float*)d_in[0];
  const float* k  = (const float*)d_in[1];
  const float* v  = (const float*)d_in[2];
  const float* Wq = (const float*)d_in[3];
  const float* bq = (const float*)d_in[4];
  const float* Wk = (const float*)d_in[5];
  const float* bk = (const float*)d_in[6];
  const float* Wv = (const float*)d_in[7];
  const float* bv = (const float*)d_in[8];

  float* outputs = (float*)d_out;                      // [B,S,D]
  float* scores  = outputs + (size_t)B * S * D;        // [B,S,S]

  // Workspace carve-up (all chunks 256B-aligned by construction)
  char* ws = (char*)d_ws;
  size_t off = 0;
  _Float16* W16q = (_Float16*)(ws + off); off += (size_t)D * D * 2;
  _Float16* W16k = (_Float16*)(ws + off); off += (size_t)D * D * 2;
  _Float16* W16v = (_Float16*)(ws + off); off += (size_t)D * D * 2;
  _Float16* qh   = (_Float16*)(ws + off); off += (size_t)M * D * 2;  // [M,D]
  _Float16* kh   = (_Float16*)(ws + off); off += (size_t)M * D * 2;  // [M,D]
  _Float16* vhT  = (_Float16*)(ws + off); off += (size_t)B * D * S * 2; // [B,D,S]
  _Float16* prob = (_Float16*)(ws + off); off += (size_t)B * S * S * 2; // [B,S,S]

  // 1) Weights -> f16
  {
    int n = D * D, blk = 256, g = (n + blk - 1) / blk;
    cvt_f32_to_f16<<<g, blk, 0, stream>>>(Wq, W16q, n);
    cvt_f32_to_f16<<<g, blk, 0, stream>>>(Wk, W16k, n);
    cvt_f32_to_f16<<<g, blk, 0, stream>>>(Wv, W16v, n);
  }

  // 2) Projections (M x D) = X (M x D) * W^T + b
  {
    dim3 g(M / 128, D / 64, 1);
    proj_gemm<<<g, 256, 0, stream>>>(q, W16q, bq, qh, nullptr, D, D, S);
    proj_gemm<<<g, 256, 0, stream>>>(k, W16k, bk, kh, nullptr, D, D, S);
    proj_gemm<<<g, 256, 0, stream>>>(v, W16v, bv, nullptr, vhT, D, D, S);
  }

  // 3) scores[b] = (qh_b * kh_b^T) / sqrt(D)  -> f32 in d_out
  {
    const float scale = 1.0f / sqrtf((float)D);
    dim3 g(S / 256, S / 64, B);
    gemm_a16_bt16<<<g, 256, 0, stream>>>(qh, (size_t)S * D, D,
                                         kh, (size_t)S * D, D,
                                         scores, (size_t)S * S, S, D, scale);
  }

  // 4) softmax over last dim -> f16 probs
  softmax_rows<<<B * S, 256, 0, stream>>>(scores, prob, S);

  // 5) outputs[b] = probs_b * vh_b  (B-frags from vh^T) -> f32 in d_out
  {
    dim3 g(S / 256, D / 64, B);
    gemm_a16_bt16<<<g, 256, 0, stream>>>(prob, (size_t)S * S, S,
                                         vhT, (size_t)D * S, S,
                                         outputs, (size_t)S * D, D, S, 1.0f);
  }
}
